// MultiAgentDRQN_43705587204314
// MI455X (gfx1250) — matrix operand order repaired and checked by reading, not compiled, not run on hardware
//
#include <hip/hip_runtime.h>

#define TSTEPS 200
#define NAG 8
#define OBS 128
#define OUTD 16

typedef __attribute__((ext_vector_type(16))) _Float16 v16h;
typedef __attribute__((ext_vector_type(8)))  float    v8f;

#if __has_builtin(__builtin_amdgcn_tanhf)
  #define FAST_TANH(x) __builtin_amdgcn_tanhf(x)
#elif __has_builtin(__builtin_amdgcn_tanh_f32)
  #define FAST_TANH(x) __builtin_amdgcn_tanh_f32(x)
#else
  #define FAST_TANH(x) tanhf(x)
#endif

__device__ __forceinline__ v8f wmma16(v16h a, v16h b, v8f c) {
  // D = A(16x32 f16) * B(32x16 f16) + C(16x16 f32)
  return __builtin_amdgcn_wmma_f32_16x16x32_f16(false, a, false, b, (short)0, c,
                                                false, false);
}

__device__ __forceinline__ v8f zero8() {
  v8f z;
#pragma unroll
  for (int j = 0; j < 8; ++j) z[j] = 0.0f;
  return z;
}

// sigmoid(x) = 0.5*tanh(0.5x)+0.5 : one hardware TRANS op instead of exp+rcp chain
__device__ __forceinline__ float sigm_t(float x) {
  return 0.5f * FAST_TANH(0.5f * x) + 0.5f;
}

// A-fragment (16x32 MxK) from LDS row-major [16][rowStride] f16.
// Lane L (lo half) holds row m=L&15, K = kb+0..15; hi half kb+=16.
__device__ __forceinline__ v16h load_a_lds(const _Float16* base, int rowStride,
                                           int lane, int kstep) {
  int m  = lane & 15;
  int kb = kstep * 32 + ((lane & 16) ? 16 : 0);
  const _Float16* p = base + m * rowStride + kb;
  v16h r;
#pragma unroll
  for (int j = 0; j < 16; ++j) r[j] = p[j];
  return r;
}

// B-fragment (32x16 KxN) with B = W^T, W row-major [fanout][fanin] f16 in LDS.
__device__ __forceinline__ v16h load_b_lds(const _Float16* W, int fanin,
                                           int ncolBase, int lane, int kstep) {
  int n  = ncolBase + (lane & 15);
  int kb = kstep * 32 + ((lane & 16) ? 16 : 0);
  const _Float16* p = W + n * fanin + kb;
  v16h r;
#pragma unroll
  for (int j = 0; j < 16; ++j) r[j] = p[j];
  return r;
}

__global__ __launch_bounds__(128, 1) void drqn_gru_wmma_kernel(
    const float* __restrict__ obs,   // [B,T,N,128]
    const float* __restrict__ W1,    // [64,136]
    const float* __restrict__ b1,    // [64]
    const float* __restrict__ Wih,   // [192,64]
    const float* __restrict__ Whh,   // [192,64]
    const float* __restrict__ bih,   // [192]
    const float* __restrict__ bhh,   // [192]
    const float* __restrict__ W2,    // [16,64]
    const float* __restrict__ b2,    // [16]
    float* __restrict__ out)         // nn_out[6553600] | maxv[409600] | maxa[409600]
{
  __shared__ _Float16 sW1[64 * 128];    // fc1 weights (obs part), f16
  __shared__ _Float16 sWih[192 * 64];
  __shared__ _Float16 sWhh[192 * 64];
  __shared__ _Float16 sW2[16 * 64];
  __shared__ float    sB1[8 * 64];      // b1 + one-hot column of W1 per agent
  __shared__ float    sBih[192], sBhh[192], sB2v[16];
  __shared__ _Float16 sX[16 * 128];     // staged x tile (f16), shared by 4 waves
  __shared__ _Float16 sA[16 * 64];      // fc1 activation
  __shared__ _Float16 sH[16 * 64];      // hidden state

  const int tid  = threadIdx.x;
  const int wave = tid >> 5;
  const int lane = tid & 31;

  // ---- stage weights (f32 -> f16) into LDS; fold one-hot into per-agent bias ----
  for (int idx = tid; idx < 64 * 136; idx += 128) {
    int j = idx / 136, k = idx % 136;
    float v = W1[idx];
    if (k < 128) sW1[j * 128 + k] = (_Float16)v;
    else         sB1[(k - 128) * 64 + j] = b1[j] + v;   // agent one-hot column
  }
  for (int idx = tid; idx < 192 * 64; idx += 128) {
    sWih[idx] = (_Float16)Wih[idx];
    sWhh[idx] = (_Float16)Whh[idx];
  }
  for (int idx = tid; idx < 16 * 64; idx += 128) sW2[idx] = (_Float16)W2[idx];
  for (int idx = tid; idx < 192; idx += 128) { sBih[idx] = bih[idx]; sBhh[idx] = bhh[idx]; }
  if (tid < 16) sB2v[tid] = b2[tid];
  for (int idx = tid; idx < 16 * 64; idx += 128) sH[idx] = (_Float16)0.0f;
  __syncthreads();

  const int tile = blockIdx.x;                     // 128 tiles of 16 rows
  // tile rows 0..7  -> batch 2*tile,   agents 0..7 (contiguous 4KB in obs per t)
  // tile rows 8..15 -> batch 2*tile+1, agents 0..7
  const float* c0 = obs + ((size_t)(2 * tile)     * TSTEPS) * (NAG * OBS);
  const float* c1 = obs + ((size_t)(2 * tile + 1) * TSTEPS) * (NAG * OBS);

  v8f hC = zero8();                                // this wave's h columns (c = wave)

  for (int t = 0; t < TSTEPS; ++t) {
    const float* p0 = c0 + (size_t)t * (NAG * OBS);
    const float* p1 = c1 + (size_t)t * (NAG * OBS);

    // ---- stage x tile into LDS (f32 -> f16), fully coalesced ----
#pragma unroll
    for (int j = 0; j < 8; ++j) sX[tid + 128 * j]        = (_Float16)p0[tid + 128 * j];
#pragma unroll
    for (int j = 0; j < 8; ++j) sX[1024 + tid + 128 * j] = (_Float16)p1[tid + 128 * j];

    // Prefetch upcoming timesteps. locality=3 -> near (WGP) scope: pulls t+1's
    // 4KB chunks into all cache levels of this WGP; t+2 goes to a farther level.
    {
      const float* pf = (tid < 32) ? p0 + 1024 : p1 + 1024;      // t+1
      __builtin_prefetch(pf + (tid & 31) * 32, 0, 3);
    }
    if (tid >= 64) {
      const float* pf2 = (tid < 96) ? p0 + 2048 : p1 + 2048;     // t+2
      __builtin_prefetch(pf2 + (tid & 31) * 32, 0, 1);
    }
    __syncthreads();

    // ---- fc1: wave w computes N-tile nt = w of a = relu(x @ W1^T + effbias) ----
    {
      v8f acc = zero8();
#pragma unroll
      for (int s = 0; s < 4; ++s)
        acc = wmma16(load_a_lds(sX, 128, lane, s),
                     load_b_lds(sW1, 128, wave * 16, lane, s), acc);
      int col = wave * 16 + (lane & 15);
#pragma unroll
      for (int i = 0; i < 8; ++i) {
        // row m = i or i+8 -> agent id = m&7 = i
        float v = acc[i] + sB1[i * 64 + col];
        int m = ((lane & 16) ? 8 : 0) + i;
        sA[m * 64 + col] = (_Float16)(v > 0.0f ? v : 0.0f);
      }
    }
    __syncthreads();

    // ---- GRU: wave w owns hidden-column tile c = w ----
    {
      v16h aA0 = load_a_lds(sA, 64, lane, 0), aA1 = load_a_lds(sA, 64, lane, 1);
      v16h aH0 = load_a_lds(sH, 64, lane, 0), aH1 = load_a_lds(sH, 64, lane, 1);
      const int c = wave;

      v8f giR = zero8(), ghR = zero8(), giZ = zero8(), ghZ = zero8(), giN = zero8(), ghN = zero8();
      giR = wmma16(aA0, load_b_lds(sWih, 64, c * 16, lane, 0), giR);
      giR = wmma16(aA1, load_b_lds(sWih, 64, c * 16, lane, 1), giR);
      ghR = wmma16(aH0, load_b_lds(sWhh, 64, c * 16, lane, 0), ghR);
      ghR = wmma16(aH1, load_b_lds(sWhh, 64, c * 16, lane, 1), ghR);
      giZ = wmma16(aA0, load_b_lds(sWih, 64, 64 + c * 16, lane, 0), giZ);
      giZ = wmma16(aA1, load_b_lds(sWih, 64, 64 + c * 16, lane, 1), giZ);
      ghZ = wmma16(aH0, load_b_lds(sWhh, 64, 64 + c * 16, lane, 0), ghZ);
      ghZ = wmma16(aH1, load_b_lds(sWhh, 64, 64 + c * 16, lane, 1), ghZ);
      giN = wmma16(aA0, load_b_lds(sWih, 64, 128 + c * 16, lane, 0), giN);
      giN = wmma16(aA1, load_b_lds(sWih, 64, 128 + c * 16, lane, 1), giN);
      ghN = wmma16(aH0, load_b_lds(sWhh, 64, 128 + c * 16, lane, 0), ghN);
      ghN = wmma16(aH1, load_b_lds(sWhh, 64, 128 + c * 16, lane, 1), ghN);

      int col = lane & 15;
      int cr  = c * 16 + col;
      float brr  = sBih[cr] + sBhh[cr];
      float bzz  = sBih[64 + cr] + sBhh[64 + cr];
      float bin_ = sBih[128 + cr], bhn = sBhh[128 + cr];

      float hnew[8];
#pragma unroll
      for (int i = 0; i < 8; ++i) {
        float r  = sigm_t(giR[i] + ghR[i] + brr);
        float z  = sigm_t(giZ[i] + ghZ[i] + bzz);
        float nn = FAST_TANH(giN[i] + bin_ + r * (ghN[i] + bhn));
        float h  = (1.0f - z) * nn + z * hC[i];
        hC[i] = h;
        hnew[i] = h;
      }
      __syncthreads();   // all waves finished reading old sH / sA
#pragma unroll
      for (int i = 0; i < 8; ++i) {
        int m = ((lane & 16) ? 8 : 0) + i;
        sH[m * 64 + cr] = (_Float16)hnew[i];
      }
    }
    __syncthreads();     // sH fully updated

    // ---- fc2: q = h_new @ W2^T + b2 (computed by every wave; epilogue split) ----
    {
      v16h q0 = load_a_lds(sH, 64, lane, 0), q1 = load_a_lds(sH, 64, lane, 1);
      v8f accQ = zero8();
      accQ = wmma16(q0, load_b_lds(sW2, 64, 0, lane, 0), accQ);
      accQ = wmma16(q1, load_b_lds(sW2, 64, 0, lane, 1), accQ);

      int o = lane & 15;
      float bq = sB2v[o];

      float qv[2];
      switch (wave) {              // compile-time vector extracts
        case 0:  qv[0] = accQ[0]; qv[1] = accQ[1]; break;
        case 1:  qv[0] = accQ[2]; qv[1] = accQ[3]; break;
        case 2:  qv[0] = accQ[4]; qv[1] = accQ[5]; break;
        default: qv[0] = accQ[6]; qv[1] = accQ[7]; break;
      }

#pragma unroll
      for (int ii = 0; ii < 2; ++ii) {
        int i  = wave * 2 + ii;
        int m  = ((lane & 16) ? 8 : 0) + i;
        int gr = tile * 16 + m;
        int rowq = ((gr >> 3) * TSTEPS + t) * NAG + (gr & 7);   // (b,t,n) flat
        float q = qv[ii] + bq;
        out[rowq * OUTD + o] = q;

        // row max / argmax over 16 outputs: butterfly within each 16-lane half
        float mv = q;
        int   mi = o;
#pragma unroll
        for (int mask = 1; mask < 16; mask <<= 1) {
          float ov = __shfl_xor(mv, mask, 32);
          int   oi = __shfl_xor(mi, mask, 32);
          if (ov > mv || (ov == mv && oi < mi)) { mv = ov; mi = oi; }
        }
        if (o == 0) {
          out[6553600 + rowq]          = mv;
          out[6553600 + 409600 + rowq] = (float)mi;
        }
      }
    }
  }
}

extern "C" void kernel_launch(void* const* d_in, const int* in_sizes, int n_in,
                              void* d_out, int out_size, void* d_ws, size_t ws_size,
                              hipStream_t stream) {
  (void)in_sizes; (void)n_in; (void)d_ws; (void)ws_size; (void)out_size;
  // inputs: 0 obs_history, 1 agent(int32, ids 0..7 per slot -> folded analytically),
  //         2 W1, 3 b1, 4 W_ih, 5 W_hh, 6 b_ih, 7 b_hh, 8 W2, 9 b2
  drqn_gru_wmma_kernel<<<128, 128, 0, stream>>>(
      (const float*)d_in[0],
      (const float*)d_in[2], (const float*)d_in[3],
      (const float*)d_in[4], (const float*)d_in[5],
      (const float*)d_in[6], (const float*)d_in[7],
      (const float*)d_in[8], (const float*)d_in[9],
      (float*)d_out);
}
